// MultiHeadCausalSelfAttention_38809324486806
// MI455X (gfx1250) — matrix operand order repaired and checked by reading, not compiled, not run on hardware
//
#include <hip/hip_runtime.h>
#include <hip/hip_bf16.h>

typedef __attribute__((ext_vector_type(16))) __bf16 v16bf;
typedef __attribute__((ext_vector_type(8)))  __bf16 v8bf;
typedef __attribute__((ext_vector_type(8)))  float  v8f;

#define DM 1024
#define TT 2048
#define NH 16
#define HD 64

// ---------------------------------------------------------------------------
// gfx1250 async global->LDS copy (ASYNCcnt path). LDS destination address is
// the low 32 bits of the flat shared-memory pointer;
// global address in a VGPR pair, GV mode.
// ---------------------------------------------------------------------------
__device__ __forceinline__ void async_ld_b128(void* lds_dst, const void* src) {
    unsigned int lo = (unsigned int)(unsigned long long)lds_dst;
    asm volatile("global_load_async_to_lds_b128 %0, %1, off"
                 :: "v"(lo), "v"(src) : "memory");
}
__device__ __forceinline__ void wait_async0() {
    asm volatile("s_wait_asynccnt 0" ::: "memory");
}

// ---------------------------------------------------------------------------
// WMMA fragment loaders (CDNA5 16x16x32 bf16 layouts, wave32)
// A (16x32): lane L holds row L%16; element i <-> K = 16*(i/8) + 8*(L/16) + i%8
// B (32x16): lane L holds col L%16; element i <-> K = 16*(i/8) + 8*(L/16) + i%8
//            -> loaded from an [N][K] ("Bt") tile: fully contiguous
// C/D (16x16 f32): vgpr r <-> row r + 8*(L/16), col L%16
// ---------------------------------------------------------------------------
__device__ __forceinline__ v16bf load_a_frag(const __bf16* base, int row, int ld,
                                             int koff, int lane) {
    int g = (lane >> 4) & 1;
    const __bf16* p = base + row * ld + koff;
    v8bf lo = *(const v8bf*)(p + 8 * g);
    v8bf hi = *(const v8bf*)(p + 16 + 8 * g);
    v16bf r;
#pragma unroll
    for (int i = 0; i < 8; ++i) { r[i] = lo[i]; r[i + 8] = hi[i]; }
    return r;
}

__device__ __forceinline__ v16bf load_b_frag(const __bf16* base, int n, int ld,
                                             int koff, int lane) {
    int g = (lane >> 4) & 1;
    const __bf16* p = base + n * ld + koff + 8 * g;
    v8bf lo = *(const v8bf*)(p);
    v8bf hi = *(const v8bf*)(p + 16);
    v16bf r;
#pragma unroll
    for (int i = 0; i < 8; ++i) { r[i] = lo[i]; r[i + 8] = hi[i]; }
    return r;
}

__device__ __forceinline__ v8f wmma_bf16(v16bf a, v16bf b, v8f c) {
    return __builtin_amdgcn_wmma_f32_16x16x32_bf16(false, a, false, b,
                                                   (short)0, c, false, false);
}

// ---------------------------------------------------------------------------
// Conversion kernels
// ---------------------------------------------------------------------------
__global__ void cvt_bf16(const float* __restrict__ in, __bf16* __restrict__ out, int n) {
    for (int i = blockIdx.x * blockDim.x + threadIdx.x; i < n;
         i += gridDim.x * blockDim.x)
        out[i] = (__bf16)in[i];
}

// in: [K][N] f32  ->  out: [N][K] bf16
__global__ void cvt_tr_bf16(const float* __restrict__ in, __bf16* __restrict__ out,
                            int K, int N) {
    int total = K * N;
    for (int i = blockIdx.x * blockDim.x + threadIdx.x; i < total;
         i += gridDim.x * blockDim.x) {
        int n = i / K, k = i % K;
        out[i] = (__bf16)in[k * N + n];
    }
}

// ---------------------------------------------------------------------------
// GEMM 1: qkv = X @ Wqkv + b ; epilogue scatters Q (scaled), K, V^T as bf16
// Block: 256 thr (8 waves), C tile 128x128, K-step 32, double-buffered LDS
// fed by async global->LDS copies.
// ---------------------------------------------------------------------------
__global__ __launch_bounds__(256)
void gemm_qkv(const __bf16* __restrict__ A,   // [4096][1024]
              const __bf16* __restrict__ Bt,  // [3072][1024]
              const float* __restrict__ bias, // [3072]
              __bf16* __restrict__ Q, __bf16* __restrict__ K,
              __bf16* __restrict__ Vt) {
    __shared__ alignas(16) __bf16 As[2][128 * 32];
    __shared__ alignas(16) __bf16 Bs[2][128 * 32];
    const int tid = threadIdx.x, lane = tid & 31, wave = tid >> 5;
    const int wm = wave >> 1, wn = wave & 1, l16 = lane & 15, g = (lane >> 4) & 1;
    const int m0 = blockIdx.y * 128, n0 = blockIdx.x * 128;

    v8f acc[2][4] = {};

    // prologue: fetch tile 0
    {
#pragma unroll
        for (int i = 0; i < 2; ++i) {
            int ch = tid + i * 256;
            int row = ch >> 2, seg = ch & 3;
            async_ld_b128(&As[0][row * 32 + seg * 8],
                          &A[(size_t)(m0 + row) * DM + seg * 8]);
            async_ld_b128(&Bs[0][row * 32 + seg * 8],
                          &Bt[(size_t)(n0 + row) * DM + seg * 8]);
        }
        wait_async0();
        __syncthreads();
    }

    int buf = 0;
    for (int k0 = 0; k0 < DM; k0 += 32) {
        if (k0 + 32 < DM) {  // prefetch next tile into the other buffer
#pragma unroll
            for (int i = 0; i < 2; ++i) {
                int ch = tid + i * 256;
                int row = ch >> 2, seg = ch & 3;
                async_ld_b128(&As[buf ^ 1][row * 32 + seg * 8],
                              &A[(size_t)(m0 + row) * DM + k0 + 32 + seg * 8]);
                async_ld_b128(&Bs[buf ^ 1][row * 32 + seg * 8],
                              &Bt[(size_t)(n0 + row) * DM + k0 + 32 + seg * 8]);
            }
        }

        v16bf af[2], bfr[4];
#pragma unroll
        for (int mt = 0; mt < 2; ++mt)
            af[mt] = load_a_frag(As[buf], wm * 32 + mt * 16 + l16, 32, 0, lane);
#pragma unroll
        for (int nt = 0; nt < 4; ++nt)
            bfr[nt] = load_b_frag(Bs[buf], wn * 64 + nt * 16 + l16, 32, 0, lane);
#pragma unroll
        for (int mt = 0; mt < 2; ++mt)
#pragma unroll
            for (int nt = 0; nt < 4; ++nt)
                acc[mt][nt] = wmma_bf16(af[mt], bfr[nt], acc[mt][nt]);

        wait_async0();
        __syncthreads();
        buf ^= 1;
    }

#pragma unroll
    for (int mt = 0; mt < 2; ++mt)
#pragma unroll
        for (int nt = 0; nt < 4; ++nt)
#pragma unroll
            for (int r = 0; r < 8; ++r) {
                int row = m0 + wm * 32 + mt * 16 + r + 8 * g;
                int col = n0 + wn * 64 + nt * 16 + l16;
                float v = acc[mt][nt][r] + bias[col];
                int t = row & (TT - 1), b = row >> 11;
                int which = col >> 10, c = col & (DM - 1);
                int h = c >> 6, d = c & (HD - 1);
                size_t bh = (size_t)(b * NH + h);
                if (which == 0)
                    Q[(bh * TT + t) * HD + d] = (__bf16)(v * 0.125f);  // fold 1/sqrt(64)
                else if (which == 1)
                    K[(bh * TT + t) * HD + d] = (__bf16)v;
                else
                    Vt[(bh * HD + d) * TT + t] = (__bf16)v;           // transposed V
            }
}

// ---------------------------------------------------------------------------
// Flash attention: grid (B*H, T/64), 128 thr (4 waves), 64-query tile.
// K/V tiles streamed with async global->LDS copies.
// ---------------------------------------------------------------------------
__global__ __launch_bounds__(128)
void flash_attn(const __bf16* __restrict__ Q, const __bf16* __restrict__ Kg,
                const __bf16* __restrict__ Vt, __bf16* __restrict__ O) {
    __shared__ alignas(16) __bf16 Qs[64 * 64];
    __shared__ alignas(16) __bf16 Ks[64 * 64];
    __shared__ alignas(16) __bf16 Vs[64 * 64];  // V^T tile: [d][s]
    __shared__ alignas(16) __bf16 Ps[4][16 * 64];

    const int tid = threadIdx.x, lane = tid & 31, wave = tid >> 5;
    const int l16 = lane & 15, g = (lane >> 4) & 1;
    const int bh = blockIdx.x, q0 = blockIdx.y * 64;
    const int b = bh >> 4, h = bh & (NH - 1);

    const __bf16* Qh = Q + (size_t)bh * TT * HD;
    const __bf16* Kh = Kg + (size_t)bh * TT * HD;
    const __bf16* Vh = Vt + (size_t)bh * HD * TT;

#pragma unroll
    for (int i = 0; i < 4; ++i) {  // Q tile once per block (async)
        int ch = tid + i * 128;
        int row = ch >> 3, seg = ch & 7;
        async_ld_b128(&Qs[row * 64 + seg * 8],
                      &Qh[(size_t)(q0 + row) * HD + seg * 8]);
    }

    v8f o[4] = {};
    float mrun[8], lrun[8];
#pragma unroll
    for (int r = 0; r < 8; ++r) { mrun[r] = -__builtin_inff(); lrun[r] = 0.f; }

    for (int s0 = 0; s0 <= q0; s0 += 64) {
#pragma unroll
        for (int i = 0; i < 4; ++i) {
            int ch = tid + i * 128;
            int row = ch >> 3, seg = ch & 7;
            async_ld_b128(&Ks[row * 64 + seg * 8],
                          &Kh[(size_t)(s0 + row) * HD + seg * 8]);
            async_ld_b128(&Vs[row * 64 + seg * 8],
                          &Vh[(size_t)row * TT + s0 + seg * 8]);
        }
        wait_async0();
        __syncthreads();  // also covers the initial async Q load on iter 0

        // S = Q · K^T  (Ks rows [key][d] are exactly the Bt layout)
        v16bf aq0 = load_a_frag(Qs, wave * 16 + l16, 64, 0, lane);
        v16bf aq1 = load_a_frag(Qs, wave * 16 + l16, 64, 32, lane);
        v8f s[4] = {};
#pragma unroll
        for (int nt = 0; nt < 4; ++nt) {
            v16bf bb = load_b_frag(Ks, nt * 16 + l16, 64, 0, lane);
            s[nt] = wmma_bf16(aq0, bb, s[nt]);
            bb = load_b_frag(Ks, nt * 16 + l16, 64, 32, lane);
            s[nt] = wmma_bf16(aq1, bb, s[nt]);
        }

        if (s0 == q0) {  // causal mask, diagonal tile only
#pragma unroll
            for (int nt = 0; nt < 4; ++nt)
#pragma unroll
                for (int r = 0; r < 8; ++r) {
                    int col = s0 + nt * 16 + l16;
                    int rowq = q0 + wave * 16 + r + 8 * g;
                    if (col > rowq) s[nt][r] = -__builtin_inff();
                }
        }

        // online softmax (row stats across 16-lane halves, wave32 shfl_xor)
        float mnew[8], alpha[8];
#pragma unroll
        for (int r = 0; r < 8; ++r) {
            float mx = s[0][r];
#pragma unroll
            for (int nt = 1; nt < 4; ++nt) mx = fmaxf(mx, s[nt][r]);
            mx = fmaxf(mx, __shfl_xor(mx, 1));
            mx = fmaxf(mx, __shfl_xor(mx, 2));
            mx = fmaxf(mx, __shfl_xor(mx, 4));
            mx = fmaxf(mx, __shfl_xor(mx, 8));
            float mn = fmaxf(mrun[r], mx);
            alpha[r] = __expf(mrun[r] - mn);
            mrun[r] = mn;
            mnew[r] = mn;
        }
#pragma unroll
        for (int nt = 0; nt < 4; ++nt)
#pragma unroll
            for (int r = 0; r < 8; ++r)
                s[nt][r] = __expf(s[nt][r] - mnew[r]);
#pragma unroll
        for (int r = 0; r < 8; ++r) {
            float rs = s[0][r] + s[1][r] + s[2][r] + s[3][r];
            rs += __shfl_xor(rs, 1);
            rs += __shfl_xor(rs, 2);
            rs += __shfl_xor(rs, 4);
            rs += __shfl_xor(rs, 8);
            lrun[r] = lrun[r] * alpha[r] + rs;
        }
#pragma unroll
        for (int nt = 0; nt < 4; ++nt)
#pragma unroll
            for (int r = 0; r < 8; ++r) o[nt][r] *= alpha[r];

        // stage P through LDS: C-layout -> A-layout
        __bf16* Pw = &Ps[wave][0];
#pragma unroll
        for (int nt = 0; nt < 4; ++nt)
#pragma unroll
            for (int r = 0; r < 8; ++r)
                Pw[(r + 8 * g) * 64 + nt * 16 + l16] = (__bf16)s[nt][r];
        asm volatile("s_wait_dscnt 0" ::: "memory");  // wave-local LDS RAW fence

        // O += P · V  (Vs rows [d][s] are the Bt layout for B = V)
        v16bf ap0 = load_a_frag(Pw, l16, 64, 0, lane);
        v16bf ap1 = load_a_frag(Pw, l16, 64, 32, lane);
#pragma unroll
        for (int nt = 0; nt < 4; ++nt) {
            v16bf bb = load_b_frag(Vs, nt * 16 + l16, 64, 0, lane);
            o[nt] = wmma_bf16(ap0, bb, o[nt]);
            bb = load_b_frag(Vs, nt * 16 + l16, 64, 32, lane);
            o[nt] = wmma_bf16(ap1, bb, o[nt]);
        }
        __syncthreads();  // before next iteration overwrites Ks/Vs
    }

#pragma unroll
    for (int nt = 0; nt < 4; ++nt)
#pragma unroll
        for (int r = 0; r < 8; ++r) {
            int qrow = q0 + wave * 16 + r + 8 * g;
            int d = nt * 16 + l16;
            float v = o[nt][r] / lrun[r];
            O[(size_t)(b * TT + qrow) * DM + h * HD + d] = (__bf16)v;
        }
}

// ---------------------------------------------------------------------------
// GEMM 2: out = Attn @ Wout + b_out  -> f32 (same pipelined structure)
// ---------------------------------------------------------------------------
__global__ __launch_bounds__(256)
void gemm_out(const __bf16* __restrict__ A,   // [4096][1024]
              const __bf16* __restrict__ Bt,  // [1024][1024]
              const float* __restrict__ bias, float* __restrict__ out) {
    __shared__ alignas(16) __bf16 As[2][128 * 32];
    __shared__ alignas(16) __bf16 Bs[2][128 * 32];
    const int tid = threadIdx.x, lane = tid & 31, wave = tid >> 5;
    const int wm = wave >> 1, wn = wave & 1, l16 = lane & 15, g = (lane >> 4) & 1;
    const int m0 = blockIdx.y * 128, n0 = blockIdx.x * 128;

    v8f acc[2][4] = {};

    {
#pragma unroll
        for (int i = 0; i < 2; ++i) {
            int ch = tid + i * 256;
            int row = ch >> 2, seg = ch & 3;
            async_ld_b128(&As[0][row * 32 + seg * 8],
                          &A[(size_t)(m0 + row) * DM + seg * 8]);
            async_ld_b128(&Bs[0][row * 32 + seg * 8],
                          &Bt[(size_t)(n0 + row) * DM + seg * 8]);
        }
        wait_async0();
        __syncthreads();
    }

    int buf = 0;
    for (int k0 = 0; k0 < DM; k0 += 32) {
        if (k0 + 32 < DM) {
#pragma unroll
            for (int i = 0; i < 2; ++i) {
                int ch = tid + i * 256;
                int row = ch >> 2, seg = ch & 3;
                async_ld_b128(&As[buf ^ 1][row * 32 + seg * 8],
                              &A[(size_t)(m0 + row) * DM + k0 + 32 + seg * 8]);
                async_ld_b128(&Bs[buf ^ 1][row * 32 + seg * 8],
                              &Bt[(size_t)(n0 + row) * DM + k0 + 32 + seg * 8]);
            }
        }

        v16bf af[2], bfr[4];
#pragma unroll
        for (int mt = 0; mt < 2; ++mt)
            af[mt] = load_a_frag(As[buf], wm * 32 + mt * 16 + l16, 32, 0, lane);
#pragma unroll
        for (int nt = 0; nt < 4; ++nt)
            bfr[nt] = load_b_frag(Bs[buf], wn * 64 + nt * 16 + l16, 32, 0, lane);
#pragma unroll
        for (int mt = 0; mt < 2; ++mt)
#pragma unroll
            for (int nt = 0; nt < 4; ++nt)
                acc[mt][nt] = wmma_bf16(af[mt], bfr[nt], acc[mt][nt]);

        wait_async0();
        __syncthreads();
        buf ^= 1;
    }

#pragma unroll
    for (int mt = 0; mt < 2; ++mt)
#pragma unroll
        for (int nt = 0; nt < 4; ++nt)
#pragma unroll
            for (int r = 0; r < 8; ++r) {
                int row = m0 + wm * 32 + mt * 16 + r + 8 * g;
                int col = n0 + wn * 64 + nt * 16 + l16;
                out[(size_t)row * DM + col] = acc[mt][nt][r] + bias[col];
            }
}

// ---------------------------------------------------------------------------
extern "C" void kernel_launch(void* const* d_in, const int* in_sizes, int n_in,
                              void* d_out, int out_size, void* d_ws, size_t ws_size,
                              hipStream_t stream) {
    (void)in_sizes; (void)n_in; (void)out_size; (void)ws_size;
    const float* x     = (const float*)d_in[0];  // [2,2048,1024]
    const float* w_qkv = (const float*)d_in[1];  // [1024,3072]
    const float* b_qkv = (const float*)d_in[2];  // [3072]
    const float* w_out = (const float*)d_in[3];  // [1024,1024]
    const float* b_out = (const float*)d_in[4];  // [1024]
    float* out = (float*)d_out;

    char* ws = (char*)d_ws;
    size_t off = 0;
    auto walloc = [&](size_t nelem) -> __bf16* {
        __bf16* p = (__bf16*)(ws + off);
        off += (nelem * sizeof(__bf16) + 255) & ~(size_t)255;
        return p;
    };
    const size_t NT = 2ull * TT;               // 4096 rows
    __bf16* xb    = walloc(NT * DM);           // x in bf16
    __bf16* wqkvT = walloc(3ull * DM * DM);    // [3072][1024]
    __bf16* woT   = walloc((size_t)DM * DM);   // [1024][1024]
    __bf16* q     = walloc(NT * DM);           // [b,h,t,d] (pre-scaled)
    __bf16* k     = walloc(NT * DM);           // [b,h,t,d]
    __bf16* vt    = walloc(NT * DM);           // [b,h,d,t]
    __bf16* attn  = walloc(NT * DM);           // [b*t][h*d]

    cvt_bf16<<<1024, 256, 0, stream>>>(x, xb, (int)(NT * DM));
    cvt_tr_bf16<<<1024, 256, 0, stream>>>(w_qkv, wqkvT, DM, 3 * DM);
    cvt_tr_bf16<<<512, 256, 0, stream>>>(w_out, woT, DM, DM);

    dim3 g1(3 * DM / 128, NT / 128);
    gemm_qkv<<<g1, 256, 0, stream>>>(xb, wqkvT, b_qkv, q, k, vt);

    dim3 g2(2 * NH, TT / 64);
    flash_attn<<<g2, 128, 0, stream>>>(q, k, vt, attn);

    dim3 g3(DM / 128, NT / 128);
    gemm_out<<<g3, 256, 0, stream>>>(attn, woT, b_out, out);
}
